// GNNTransformerDetectorV5_89781996356228
// MI455X (gfx1250) — matrix operand order repaired and checked by reading, not compiled, not run on hardware
//
#include <hip/hip_runtime.h>
#include <math.h>

// ---------------------------------------------------------------------------
// Types for CDNA5 WMMA (wave32, gfx1250)
// ---------------------------------------------------------------------------
typedef __attribute__((ext_vector_type(16))) __bf16          v16bf;
typedef __attribute__((ext_vector_type(16))) unsigned short  v16us;
typedef __attribute__((ext_vector_type(8)))  float           v8f;

__device__ __forceinline__ unsigned short f2bf(float f) {
  unsigned int u = __float_as_uint(f);
  u += 0x7FFFu + ((u >> 16) & 1u);          // round-to-nearest-even
  return (unsigned short)(u >> 16);
}

__device__ __forceinline__ double shfl_xor_d(double v, int m) {
  int2 p = __builtin_bit_cast(int2, v);
  p.x = __shfl_xor(p.x, m, 32);
  p.y = __shfl_xor(p.y, m, 32);
  return __builtin_bit_cast(double, p);
}
__device__ __forceinline__ double wsum_d(double v) {
  for (int m = 16; m; m >>= 1) v += shfl_xor_d(v, m);
  return v;
}

// ---------------------------------------------------------------------------
// Weight prep: f32 (K x N) row-major  ->  bf16 transposed (N x Kpad), zero pad
// ---------------------------------------------------------------------------
__global__ void prep_wt_kernel(const float* __restrict__ W, unsigned short* __restrict__ Wt,
                               int K, int N, int Kpad) {
  int n = blockIdx.x;
  for (int k = threadIdx.x; k < Kpad; k += blockDim.x) {
    float v = (k < K) ? W[(size_t)k * N + n] : 0.f;
    Wt[(size_t)n * Kpad + k] = f2bf(v);
  }
}

// ---------------------------------------------------------------------------
// Flagship bf16 WMMA GEMM:  Y(M,N) = act(Xbf(M,Kd) @ Wt^T + bias)
//   Xbf : bf16 row-major, leading dim ldx (halves)
//   Wt  : bf16, N x Kd row-major (pre-transposed weights)
//   Block = 128 threads (4 waves), tile 64x64, K-step 32.
//   Tiles are fetched with GLOBAL_LOAD_ASYNC_TO_LDS_B128 (ASYNCcnt-tracked),
//   double-buffered: next K tile in flight while WMMA consumes current one.
//   Each wave issues exactly 4 async ops per K tile; async loads retire in
//   order, so "s_wait_asynccnt 4" = current tile landed, next still in flight.
// ---------------------------------------------------------------------------
union FragU { uint4 q[2]; v16us us; };

__global__ __launch_bounds__(128)
void gemm_bf16_wmma(const unsigned short* __restrict__ X, int ldx,
                    const unsigned short* __restrict__ Wt, int Kd,
                    const float* __restrict__ bias,
                    float* __restrict__ Yf,
                    unsigned short* __restrict__ Ybf, int ldyb,
                    int Ncols, int relu) {
  __shared__ unsigned short As[2][64 * 40];   // 40-half stride: bank-conflict pad
  __shared__ unsigned short Bs[2][64 * 40];
  const int tid  = threadIdx.x;
  const int lane = tid & 31;
  const int wave = tid >> 5;
  const int m0   = blockIdx.x * 64;
  const int n0   = blockIdx.y * 64;
  const int hsel = lane >> 4;                 // 0: lanes 0-15, 1: lanes 16-31

  // per-thread cooperative-load coordinates (2 x b128 per tile per thread)
  const int r0 = tid >> 2;                    // rows 0..31 (chunk 0), +32 (chunk 1)
  const int c0 = (tid & 3) << 3;              // half-column 0/8/16/24
  const unsigned short* gA0 = X  + (size_t)(m0 + r0)      * ldx + c0;
  const unsigned short* gA1 = X  + (size_t)(m0 + r0 + 32) * ldx + c0;
  const unsigned short* gB0 = Wt + (size_t)(n0 + r0)      * Kd  + c0;
  const unsigned short* gB1 = Wt + (size_t)(n0 + r0 + 32) * Kd  + c0;
  unsigned lA0[2], lA1[2], lB0[2], lB1[2];
#pragma unroll
  for (int b = 0; b < 2; ++b) {
    lA0[b] = (unsigned)(uintptr_t)&As[b][r0 * 40 + c0];
    lA1[b] = (unsigned)(uintptr_t)&As[b][(r0 + 32) * 40 + c0];
    lB0[b] = (unsigned)(uintptr_t)&Bs[b][r0 * 40 + c0];
    lB1[b] = (unsigned)(uintptr_t)&Bs[b][(r0 + 32) * 40 + c0];
  }

  auto issue_tile = [&](int buf) {
    asm volatile("global_load_async_to_lds_b128 %0, %1, off"
                 :: "v"(lA0[buf]), "v"(gA0) : "memory");
    asm volatile("global_load_async_to_lds_b128 %0, %1, off"
                 :: "v"(lA1[buf]), "v"(gA1) : "memory");
    asm volatile("global_load_async_to_lds_b128 %0, %1, off"
                 :: "v"(lB0[buf]), "v"(gB0) : "memory");
    asm volatile("global_load_async_to_lds_b128 %0, %1, off"
                 :: "v"(lB1[buf]), "v"(gB1) : "memory");
    gA0 += 32; gA1 += 32; gB0 += 32; gB1 += 32;   // advance one K tile
  };

  v8f acc[4] = {};
  int cur = 0;
  issue_tile(0);
  for (int k0 = 0; k0 < Kd; k0 += 32) {
    if (k0 + 32 < Kd) {
      issue_tile(cur ^ 1);                        // prefetch next K tile
      asm volatile("s_wait_asynccnt 0x4" ::: "memory");
    } else {
      asm volatile("s_wait_asynccnt 0x0" ::: "memory");
    }
    __syncthreads();                              // current tile visible to all waves

    // A fragment (16x32 bf16): per CDNA5 layout, lane<16 K-base 0, lane>=16 K-base 8,
    // halves 0..7 -> K=base..base+7, halves 8..15 -> K=base+16..base+23
    FragU fa;
    const unsigned short* ap = &As[cur][(wave * 16 + (lane & 15)) * 40 + (hsel ? 8 : 0)];
    fa.q[0] = *(const uint4*)ap;
    fa.q[1] = *(const uint4*)(ap + 16);
    v16bf av = __builtin_bit_cast(v16bf, fa.us);
#pragma unroll
    for (int nt = 0; nt < 4; ++nt) {
      // B fragment (32x16 bf16): lane = N, same K order as A (lanes>=16 K-base 8; halves 8..15 -> K=base+16..base+23)
      FragU fb;
      const unsigned short* bp = &Bs[cur][(nt * 16 + (lane & 15)) * 40 + (hsel ? 8 : 0)];
      fb.q[0] = *(const uint4*)bp;
      fb.q[1] = *(const uint4*)(bp + 16);
      v16bf bv = __builtin_bit_cast(v16bf, fb.us);
      acc[nt] = __builtin_amdgcn_wmma_f32_16x16x32_bf16(false, av, false, bv,
                                                        (short)0, acc[nt], false, false);
    }
    __syncthreads();                 // all waves done reading before buffer is reused
    cur ^= 1;
  }

#pragma unroll
  for (int nt = 0; nt < 4; ++nt) {
#pragma unroll
    for (int r = 0; r < 8; ++r) {
      int row = m0 + wave * 16 + r + (hsel ? 8 : 0);   // C/D layout: VGPR r -> M=r / r+8
      int col = n0 + nt * 16 + (lane & 15);
      float v = acc[nt][r] + bias[col];
      if (relu) v = fmaxf(v, 0.f);
      if (Yf)  Yf [(size_t)row * Ncols + col] = v;
      if (Ybf) Ybf[(size_t)row * ldyb  + col] = f2bf(v);
    }
  }
}

// ---------------------------------------------------------------------------
// per-AP MMSE: 4x4 complex solve per (b,l); one wave per item, lane = subcarrier k
// ---------------------------------------------------------------------------
__global__ __launch_bounds__(256)
void per_ap_mmse_kernel(const float* __restrict__ H, const float* __restrict__ y,
                        float* __restrict__ s_init) {
  int wave = threadIdx.x >> 5, lane = threadIdx.x & 31;
  int item = blockIdx.x * 8 + wave;                  // (b*64+l), grid 512
  const float* Hb = H + (size_t)item * 256;          // (n,k,c): (n*32+k)*2+c
  const float* yb = y + (size_t)item * 8;
  double hr[4], hi[4];
#pragma unroll
  for (int n = 0; n < 4; ++n) {
    hr[n] = (double)Hb[(n * 32 + lane) * 2];
    hi[n] = (double)Hb[(n * 32 + lane) * 2 + 1];
  }
  double Ar[4][4], Ai[4][4];
#pragma unroll
  for (int n = 0; n < 4; ++n)
#pragma unroll
    for (int m = 0; m < 4; ++m) {
      Ar[n][m] = wsum_d(hr[n] * hr[m] + hi[n] * hi[m]);  // H H^H
      Ai[n][m] = wsum_d(hi[n] * hr[m] - hr[n] * hi[m]);
    }
#pragma unroll
  for (int n = 0; n < 4; ++n) Ar[n][n] += 1.0;           // NOISE_W * I
  double br[4], bi[4];
#pragma unroll
  for (int n = 0; n < 4; ++n) { br[n] = (double)yb[n * 2]; bi[n] = (double)yb[n * 2 + 1]; }
  // Gaussian elimination (SPD, no pivot), redundantly per lane
  for (int p = 0; p < 4; ++p) {
    double d  = Ar[p][p] * Ar[p][p] + Ai[p][p] * Ai[p][p];
    double pr = Ar[p][p] / d, pi = -Ai[p][p] / d;
    for (int i2 = p + 1; i2 < 4; ++i2) {
      double fr = Ar[i2][p] * pr - Ai[i2][p] * pi;
      double fi = Ar[i2][p] * pi + Ai[i2][p] * pr;
      for (int j = p; j < 4; ++j) {
        Ar[i2][j] -= fr * Ar[p][j] - fi * Ai[p][j];
        Ai[i2][j] -= fr * Ai[p][j] + fi * Ar[p][j];
      }
      double nbr = br[i2] - (fr * br[p] - fi * bi[p]);
      double nbi = bi[i2] - (fr * bi[p] + fi * br[p]);
      br[i2] = nbr; bi[i2] = nbi;
    }
  }
  double zr[4], zi[4];
  for (int p = 3; p >= 0; --p) {
    double ar2 = br[p], ai2 = bi[p];
    for (int j = p + 1; j < 4; ++j) {
      ar2 -= Ar[p][j] * zr[j] - Ai[p][j] * zi[j];
      ai2 -= Ar[p][j] * zi[j] + Ai[p][j] * zr[j];
    }
    double d = Ar[p][p] * Ar[p][p] + Ai[p][p] * Ai[p][p];
    zr[p] = (ar2 * Ar[p][p] + ai2 * Ai[p][p]) / d;
    zi[p] = (ai2 * Ar[p][p] - ar2 * Ai[p][p]) / d;
  }
  double sr = 0, si = 0;
#pragma unroll
  for (int n = 0; n < 4; ++n) {                          // conj(h) * z
    sr += hr[n] * zr[n] + hi[n] * zi[n];
    si += hr[n] * zi[n] - hi[n] * zr[n];
  }
  s_init[((size_t)item * 32 + lane) * 2]     = (float)sr;
  s_init[((size_t)item * 32 + lane) * 2 + 1] = (float)si;
}

// ---------------------------------------------------------------------------
// global MMSE: 32x32 complex SPD solve per batch b
// ---------------------------------------------------------------------------
__global__ __launch_bounds__(256)
void global_mmse_kernel(const float* __restrict__ H, const float* __restrict__ y,
                        float* __restrict__ s_glob) {
  __shared__ double Rr[32][32], Ri[32][32];
  __shared__ double rr[32], ri[32], zr[32], zi[32];
  int b = blockIdx.x, tid = threadIdx.x;
  const float* Hb = H + (size_t)b * 16384;   // (m,k,c): (m*32+k)*2+c, m = l*4+n
  const float* yb = y + (size_t)b * 512;
  for (int idx = tid; idx < 1024; idx += 256) {
    int k = idx >> 5, l2 = idx & 31;
    double sr = 0, si = 0;
    for (int m = 0; m < 256; ++m) {
      double ar  = Hb[(m * 32 + k) * 2],  ai  = Hb[(m * 32 + k) * 2 + 1];
      double br2 = Hb[(m * 32 + l2) * 2], bi2 = Hb[(m * 32 + l2) * 2 + 1];
      sr += ar * br2 + ai * bi2;             // conj(H[:,k]) . H[:,l]
      si += ar * bi2 - ai * br2;
    }
    if (k == l2) sr += 1.0;                  // NOISE_W
    Rr[k][l2] = sr; Ri[k][l2] = si;
  }
  if (tid < 32) {
    int k = tid; double sr = 0, si = 0;
    for (int m = 0; m < 256; ++m) {
      double ar = Hb[(m * 32 + k) * 2], ai = Hb[(m * 32 + k) * 2 + 1];
      double yr = yb[m * 2], yi = yb[m * 2 + 1];
      sr += ar * yr + ai * yi;
      si += ar * yi - ai * yr;
    }
    rr[k] = sr; ri[k] = si;
  }
  __syncthreads();
  for (int p = 0; p < 32; ++p) {
    if (tid > p && tid < 32) {
      int i2 = tid;
      double d  = Rr[p][p] * Rr[p][p] + Ri[p][p] * Ri[p][p];
      double fr = (Rr[i2][p] * Rr[p][p] + Ri[i2][p] * Ri[p][p]) / d;
      double fi = (Ri[i2][p] * Rr[p][p] - Rr[i2][p] * Ri[p][p]) / d;
      for (int j = p; j < 32; ++j) {
        Rr[i2][j] -= fr * Rr[p][j] - fi * Ri[p][j];
        Ri[i2][j] -= fr * Ri[p][j] + fi * Rr[p][j];
      }
      double nbr = rr[i2] - (fr * rr[p] - fi * ri[p]);
      double nbi = ri[i2] - (fr * ri[p] + fi * rr[p]);
      rr[i2] = nbr; ri[i2] = nbi;
    }
    __syncthreads();
  }
  if (tid == 0) {
    for (int p = 31; p >= 0; --p) {
      double ar2 = rr[p], ai2 = ri[p];
      for (int j = p + 1; j < 32; ++j) {
        ar2 -= Rr[p][j] * zr[j] - Ri[p][j] * zi[j];
        ai2 -= Rr[p][j] * zi[j] + Ri[p][j] * zr[j];
      }
      double d = Rr[p][p] * Rr[p][p] + Ri[p][p] * Ri[p][p];
      zr[p] = (ar2 * Rr[p][p] + ai2 * Ri[p][p]) / d;
      zi[p] = (ai2 * Rr[p][p] - ar2 * Ri[p][p]) / d;
    }
  }
  __syncthreads();
  if (tid < 32) {
    s_glob[((size_t)b * 32 + tid) * 2]     = (float)zr[tid];
    s_glob[((size_t)b * 32 + tid) * 2 + 1] = (float)zi[tid];
  }
}

// ---------------------------------------------------------------------------
// Feature builder: 4 small MLPs (hidden 32) + if_feat, writes bf16 "combined"
// (M x 160, zero-padded 134..159)
// ---------------------------------------------------------------------------
struct FeatArgs {
  const float *s_hat, *H, *bitw, *snr, *s_init, *s_glob;
  const float *mi_w1, *mi_b1, *mi_w2, *mi_b2;
  const float *de_w1, *de_b1, *de_w2, *de_b2;
  const float *gl_w1, *gl_b1, *gl_w2, *gl_b2;
  const float *ch_w1, *ch_b1, *ch_w2, *ch_b2;
  unsigned short* out;
};

__device__ void mlp2_h32(const float* in, int nin,
                         const float* w1, const float* b1,
                         const float* w2, const float* b2, float* out) {
  float hbuf[32];
  for (int j = 0; j < 32; ++j) {
    float a = b1[j];
    for (int i = 0; i < nin; ++i) a += in[i] * w1[i * 32 + j];
    hbuf[j] = fmaxf(a, 0.f);
  }
  for (int j = 0; j < 32; ++j) {
    float a = b2[j];
    for (int i = 0; i < 32; ++i) a += hbuf[i] * w2[i * 32 + j];
    out[j] = a;
  }
}

__global__ __launch_bounds__(32)
void features_kernel(FeatArgs fa) {
  int bl = blockIdx.x;                 // b*64+l
  int b = bl >> 6;
  int k = threadIdx.x;                 // subcarrier
  size_t row = (size_t)bl * 32 + k;
  const float* Hbl = fa.H + (size_t)bl * 256;
  float hp = 0.f, hf[8];
#pragma unroll
  for (int n = 0; n < 4; ++n) {
    float re = Hbl[(n * 32 + k) * 2], im = Hbl[(n * 32 + k) * 2 + 1];
    hf[2 * n] = re; hf[2 * n + 1] = im;
    hp += re * re + im * im;
  }
  float tot = hp;
  for (int m = 16; m; m >>= 1) tot += __shfl_xor(tot, m, 32);
  float interf = tot - hp;
  unsigned short* orow = fa.out + row * 160;
  float fo[32], in2[2];
  in2[0] = fa.s_init[row * 2]; in2[1] = fa.s_init[row * 2 + 1];
  mlp2_h32(in2, 2, fa.mi_w1, fa.mi_b1, fa.mi_w2, fa.mi_b2, fo);
  for (int j = 0; j < 32; ++j) orow[j] = f2bf(fo[j]);
  in2[0] = fa.s_hat[row * 2]; in2[1] = fa.s_hat[row * 2 + 1];
  mlp2_h32(in2, 2, fa.de_w1, fa.de_b1, fa.de_w2, fa.de_b2, fo);
  for (int j = 0; j < 32; ++j) orow[32 + j] = f2bf(fo[j]);
  in2[0] = fa.s_glob[((size_t)b * 32 + k) * 2];
  in2[1] = fa.s_glob[((size_t)b * 32 + k) * 2 + 1];
  mlp2_h32(in2, 2, fa.gl_w1, fa.gl_b1, fa.gl_w2, fa.gl_b2, fo);
  for (int j = 0; j < 32; ++j) orow[64 + j] = f2bf(fo[j]);
  mlp2_h32(hf, 8, fa.ch_w1, fa.ch_b1, fa.ch_w2, fa.ch_b2, fo);
  for (int j = 0; j < 32; ++j) orow[96 + j] = f2bf(fo[j]);
  orow[128] = f2bf(fa.bitw[row * 3 + 0]);
  orow[129] = f2bf(fa.bitw[row * 3 + 1]);
  orow[130] = f2bf(fa.bitw[row * 3 + 2]);
  orow[131] = f2bf(log1pf(hp));
  orow[132] = f2bf(log1pf(interf));
  orow[133] = f2bf(fa.snr[row]);
  for (int j = 134; j < 160; ++j) orow[j] = 0;
}

// ---------------------------------------------------------------------------
// BatchNorm (training stats over all M rows), LN, reductions, misc
// ---------------------------------------------------------------------------
__global__ void zero_kernel(float* p, int n) {
  int i = blockIdx.x * 256 + threadIdx.x;
  if (i < n) p[i] = 0.f;
}

__global__ __launch_bounds__(256)
void bn_stats_kernel(const float* __restrict__ x, float* __restrict__ stats, int Mrows) {
  __shared__ float s1[256], s2[256];
  int d = blockIdx.x, tid = threadIdx.x;
  float a = 0.f, b2 = 0.f;
  for (int r = tid; r < Mrows; r += 256) {
    float v = x[(size_t)r * 128 + d];
    a += v; b2 += v * v;
  }
  s1[tid] = a; s2[tid] = b2; __syncthreads();
  for (int s = 128; s > 0; s >>= 1) {
    if (tid < s) { s1[tid] += s1[tid + s]; s2[tid] += s2[tid + s]; }
    __syncthreads();
  }
  if (tid == 0) { stats[d] = s1[0]; stats[128 + d] = s2[0]; }
}

__global__ __launch_bounds__(128)
void bn_apply_kernel(const float* __restrict__ x, const float* __restrict__ stats,
                     const float* __restrict__ g, const float* __restrict__ bet,
                     float* __restrict__ hf, unsigned short* __restrict__ hbf, int Mrows) {
  int row = blockIdx.x, d = threadIdx.x;
  float inv = 1.f / (float)Mrows;
  float mu  = stats[d] * inv;
  float var = stats[128 + d] * inv - mu * mu;
  float v = (x[(size_t)row * 128 + d] - mu) * rsqrtf(var + 1e-5f) * g[d] + bet[d];
  hf [(size_t)row * 128 + d] = v;
  hbf[(size_t)row * 128 + d] = f2bf(v);
}

__global__ __launch_bounds__(128)
void resid_ln_kernel(const float* __restrict__ x, const float* __restrict__ r,
                     const float* __restrict__ g, const float* __restrict__ bet,
                     float* __restrict__ yf, unsigned short* __restrict__ ybf) {
  __shared__ float sh[128];
  int row = blockIdx.x, tid = threadIdx.x;
  float v = x[(size_t)row * 128 + tid];
  if (r) v += r[(size_t)row * 128 + tid];
  sh[tid] = v; __syncthreads();
  for (int s = 64; s > 0; s >>= 1) { if (tid < s) sh[tid] += sh[tid + s]; __syncthreads(); }
  float mean = sh[0] * (1.f / 128.f);
  __syncthreads();
  float dv = v - mean;
  sh[tid] = dv * dv; __syncthreads();
  for (int s = 64; s > 0; s >>= 1) { if (tid < s) sh[tid] += sh[tid + s]; __syncthreads(); }
  float var = sh[0] * (1.f / 128.f);
  float o = dv * rsqrtf(var + 1e-5f) * g[tid] + bet[tid];
  if (yf)  yf [(size_t)row * 128 + tid] = o;
  if (ybf) ybf[(size_t)row * 128 + tid] = f2bf(o);
}

__global__ __launch_bounds__(128)
void mean_l_kernel(const float* __restrict__ msg, float* __restrict__ mean) {
  int bk = blockIdx.x, d = threadIdx.x;        // grid B*K = 2048
  int b = bk >> 5, k = bk & 31;
  float a = 0.f;
  for (int l2 = 0; l2 < 64; ++l2)
    a += msg[((size_t)((b * 64 + l2) * 32 + k)) * 128 + d];
  mean[(size_t)bk * 128 + d] = a * (1.f / 64.f);
}

__global__ __launch_bounds__(128)
void pack_concat_kernel(const float* __restrict__ h, const float* __restrict__ mean,
                        unsigned short* __restrict__ out) {
  int row = blockIdx.x, d = threadIdx.x;       // grid M
  int b = row >> 11, k = row & 31;
  out[(size_t)row * 256 + d]       = f2bf(h[(size_t)row * 128 + d]);
  out[(size_t)row * 256 + 128 + d] = f2bf(mean[((size_t)(b * 32 + k)) * 128 + d]);
}

__global__ void agg_score_kernel(const float* __restrict__ h, const float* __restrict__ w1,
                                 const float* __restrict__ b1, const float* __restrict__ w2,
                                 const float* __restrict__ b2, float* __restrict__ score,
                                 int Mrows) {
  int row = blockIdx.x * blockDim.x + threadIdx.x;
  if (row >= Mrows) return;
  const float* x = h + (size_t)row * 128;
  float acc = b2[0];
  for (int j = 0; j < 64; ++j) {
    float a = b1[j];
    for (int i = 0; i < 128; ++i) a += x[i] * w1[i * 64 + j];
    acc += fmaxf(a, 0.f) * w2[j];
  }
  score[row] = acc;
}

__global__ __launch_bounds__(128)
void agg_softmax_u_kernel(const float* __restrict__ h, const float* __restrict__ score,
                          float* __restrict__ u) {
  __shared__ float aw[64];
  int bk = blockIdx.x, tid = threadIdx.x;      // grid 2048
  int b = bk >> 5, k = bk & 31;
  if (tid < 64) aw[tid] = score[(size_t)((b * 64 + tid) * 32) + k];
  __syncthreads();
  if (tid == 0) {
    float mx = aw[0];
    for (int l2 = 1; l2 < 64; ++l2) mx = fmaxf(mx, aw[l2]);
    float s = 0.f;
    for (int l2 = 0; l2 < 64; ++l2) { aw[l2] = __expf(aw[l2] - mx); s += aw[l2]; }
    float inv = 1.f / s;
    for (int l2 = 0; l2 < 64; ++l2) aw[l2] *= inv;
  }
  __syncthreads();
  float acc = 0.f;
  for (int l2 = 0; l2 < 64; ++l2)
    acc += aw[l2] * h[((size_t)((b * 64 + l2) * 32 + k)) * 128 + tid];
  u[(size_t)bk * 128 + tid] = acc;
}

__global__ __launch_bounds__(32)
void attention_kernel(const float* __restrict__ qkv, unsigned short* __restrict__ attnbf) {
  __shared__ float Ks[32][32], Vs[32][32];
  int b = blockIdx.x >> 2, hh = blockIdx.x & 3;  // grid B*NH = 256
  int i = threadIdx.x;                           // query index (seq = K = 32)
  const float* rowp = qkv + ((size_t)(b * 32 + i)) * 384;
  for (int d = 0; d < 32; ++d) {
    Ks[i][d] = rowp[128 + hh * 32 + d];
    Vs[i][d] = rowp[256 + hh * 32 + d];
  }
  float q[32];
  for (int d = 0; d < 32; ++d) q[d] = rowp[hh * 32 + d];
  __syncthreads();
  float sc[32]; float mx = -1e30f;
  for (int j = 0; j < 32; ++j) {
    float s = 0.f;
    for (int d = 0; d < 32; ++d) s += q[d] * Ks[j][d];
    s *= 0.17677669529663687f;                   // 1/sqrt(32)
    sc[j] = s; mx = fmaxf(mx, s);
  }
  float sum = 0.f;
  for (int j = 0; j < 32; ++j) { sc[j] = __expf(sc[j] - mx); sum += sc[j]; }
  float inv = 1.f / sum;
  for (int d = 0; d < 32; ++d) {
    float o = 0.f;
    for (int j = 0; j < 32; ++j) o += sc[j] * Vs[j][d];
    attnbf[((size_t)(b * 32 + i)) * 128 + hh * 32 + d] = f2bf(o * inv);
  }
}

__global__ void add_kernel(float* __restrict__ a, const float* __restrict__ b2, int n) {
  int i = blockIdx.x * blockDim.x + threadIdx.x;
  if (i < n) a[i] += b2[i];
}

__global__ void head_combine_kernel(const float* __restrict__ u,
                                    const float* __restrict__ w1, const float* __restrict__ b1,
                                    const float* __restrict__ w2, const float* __restrict__ b2,
                                    const float* __restrict__ alpha_p,
                                    const float* __restrict__ s_glob,
                                    float* __restrict__ out) {
  int row = blockIdx.x * blockDim.x + threadIdx.x;
  if (row >= 2048) return;
  const float* x = u + (size_t)row * 128;
  float lg0 = b2[0], lg1 = b2[1];
  for (int j = 0; j < 64; ++j) {
    float a = b1[j];
    for (int i = 0; i < 128; ++i) a += x[i] * w1[i * 64 + j];
    a = fmaxf(a, 0.f);
    lg0 += a * w2[j * 2];
    lg1 += a * w2[j * 2 + 1];
  }
  float alpha = alpha_p[0];
  float lg[2] = { lg0, lg1 };
  for (int c = 0; c < 2; ++c) {
    float sg = s_glob[(size_t)row * 2 + c];
    out[(size_t)row * 2 + c]        = lg[c];
    out[4096 + (size_t)row * 2 + c] = sg;
    out[8192 + (size_t)row * 2 + c] = sg + alpha * tanhf(lg[c]) * 0.7071067811865475f;
  }
}

// ---------------------------------------------------------------------------
// Host orchestration
// ---------------------------------------------------------------------------
extern "C" void kernel_launch(void* const* d_in, const int* in_sizes, int n_in,
                              void* d_out, int out_size, void* d_ws, size_t ws_size,
                              hipStream_t stream) {
  (void)in_sizes; (void)n_in; (void)out_size; (void)ws_size;
  const int MR = 64 * 64 * 32;   // 131072 rows (b,l,k)
  const int MU = 64 * 32;        // 2048 rows (b,k)

  // --- param leaf indices (jax tree-flatten: dict keys sorted) ---
  const int P_AGG = 5, P_ALPHA = 9, P_BN_B = 10, P_BN_G = 11;
  const int P_CH = 12, P_DE = 16, P_FU = 20, P_GL = 24;
  const int P_GNN = 28, P_HEAD = 58, P_MI = 62, P_TF = 66;
  auto F = [&](int i) { return (const float*)d_in[i]; };

  // --- workspace layout ---
  char* base = (char*)d_ws;
  size_t off = 0;
  auto alloc = [&](size_t bytes) -> char* {
    off = (off + 255) & ~(size_t)255;
    char* p = base + off; off += bytes; return p;
  };
  float* F1 = (float*)alloc((size_t)MR * 128 * 4);          // x / upd
  float* F2 = (float*)alloc((size_t)MR * 128 * 4);          // h (persistent)
  float* F3 = (float*)alloc((size_t)MR * 128 * 4);          // msg
  unsigned short* B1 = (unsigned short*)alloc((size_t)MR * 160 * 2); // combined / hbf
  unsigned short* B2 = (unsigned short*)alloc((size_t)MR * 256 * 2); // hidden-bf / concat-bf
  float* s_init = (float*)alloc((size_t)MR * 2 * 4);
  float* s_glob = (float*)alloc((size_t)MU * 2 * 4);
  float* meanb  = (float*)alloc((size_t)MU * 128 * 4);
  float* score  = (float*)alloc((size_t)MR * 4);
  float* u      = (float*)alloc((size_t)MU * 128 * 4);
  float* qkv    = (float*)alloc((size_t)MU * 384 * 4);
  float* tbuf   = (float*)alloc((size_t)MU * 128 * 4);      // wo_out / ff_out
  unsigned short* lnbf   = (unsigned short*)alloc((size_t)MU * 128 * 2);
  unsigned short* attnbf = (unsigned short*)alloc((size_t)MU * 128 * 2);
  unsigned short* ffhbf  = (unsigned short*)alloc((size_t)MU * 256 * 2);
  float* stats = (float*)alloc(256 * 4);
  unsigned short* wbf = (unsigned short*)alloc(600000 * 2);

  // --- weight prep (f32 KxN -> bf16 NxKpad transposed) ---
  size_t wpos = 0;
  auto prep = [&](int pidx, int K, int N, int Kpad) -> unsigned short* {
    unsigned short* dst = wbf + wpos; wpos += (size_t)N * Kpad;
    prep_wt_kernel<<<dim3(N), dim3(128), 0, stream>>>(F(pidx), dst, K, N, Kpad);
    return dst;
  };
  unsigned short* fu_w1t = prep(P_FU + 0, 134, 128, 160);
  unsigned short* fu_w2t = prep(P_FU + 2, 128, 128, 128);
  unsigned short *g_m1t[3], *g_m2t[3], *g_u1t[3], *g_u2t[3];
  for (int i = 0; i < 3; ++i) {
    int pb = P_GNN + 10 * i;    // +0 ln_b, +1 ln_g, +2.. msg, +6.. upd
    g_m1t[i] = prep(pb + 2, 128, 128, 128);
    g_m2t[i] = prep(pb + 4, 128, 128, 128);
    g_u1t[i] = prep(pb + 6, 256, 128, 256);
    g_u2t[i] = prep(pb + 8, 128, 128, 128);
  }
  unsigned short *t_qkvt[2], *t_wot[2], *t_f1t[2], *t_f2t[2];
  for (int t = 0; t < 2; ++t) {
    int pb = P_TF + 12 * t;     // +0 Wo, +1 Wqkv, +2 bo, +3 bqkv, +4.. ff, +8.. lns
    t_wot[t]  = prep(pb + 0, 128, 128, 128);
    t_qkvt[t] = prep(pb + 1, 128, 384, 128);
    t_f1t[t]  = prep(pb + 4, 128, 256, 128);
    t_f2t[t]  = prep(pb + 6, 256, 128, 256);
  }

  auto gemm = [&](const unsigned short* X, int ldx, const unsigned short* Wt, int Kd,
                  const float* bias, float* Yf, unsigned short* Ybf, int ldyb,
                  int Mrows, int N, int relu) {
    gemm_bf16_wmma<<<dim3(Mrows / 64, N / 64), dim3(128), 0, stream>>>(
        X, ldx, Wt, Kd, bias, Yf, Ybf, ldyb, N, relu);
  };

  // --- MMSE front-ends (f64) ---
  per_ap_mmse_kernel<<<dim3(512), dim3(256), 0, stream>>>(F(1), F(2), s_init);
  global_mmse_kernel<<<dim3(64), dim3(256), 0, stream>>>(F(1), F(2), s_glob);

  // --- feature MLPs -> combined bf16 (M x 160) ---
  FeatArgs fa;
  fa.s_hat = F(0); fa.H = F(1); fa.bitw = F(3); fa.snr = F(4);
  fa.s_init = s_init; fa.s_glob = s_glob;
  fa.mi_w1 = F(P_MI); fa.mi_b1 = F(P_MI + 1); fa.mi_w2 = F(P_MI + 2); fa.mi_b2 = F(P_MI + 3);
  fa.de_w1 = F(P_DE); fa.de_b1 = F(P_DE + 1); fa.de_w2 = F(P_DE + 2); fa.de_b2 = F(P_DE + 3);
  fa.gl_w1 = F(P_GL); fa.gl_b1 = F(P_GL + 1); fa.gl_w2 = F(P_GL + 2); fa.gl_b2 = F(P_GL + 3);
  fa.ch_w1 = F(P_CH); fa.ch_b1 = F(P_CH + 1); fa.ch_w2 = F(P_CH + 2); fa.ch_b2 = F(P_CH + 3);
  fa.out = B1;
  features_kernel<<<dim3(4096), dim3(32), 0, stream>>>(fa);

  // --- fusion MLP (WMMA) + BatchNorm ---
  gemm(B1, 160, fu_w1t, 160, F(P_FU + 1), nullptr, B2, 128, MR, 128, 1);
  gemm(B2, 128, fu_w2t, 128, F(P_FU + 3), F1, nullptr, 0, MR, 128, 0);
  zero_kernel<<<dim3(1), dim3(256), 0, stream>>>(stats, 256);
  bn_stats_kernel<<<dim3(128), dim3(256), 0, stream>>>(F1, stats, MR);
  bn_apply_kernel<<<dim3(MR), dim3(128), 0, stream>>>(F1, stats, F(P_BN_G), F(P_BN_B),
                                                      F2, B1, MR);

  // --- 3 mean-field GNN layers (WMMA) ---
  for (int i = 0; i < 3; ++i) {
    int pb = P_GNN + 10 * i;
    gemm(B1, 128, g_m1t[i], 128, F(pb + 3), nullptr, B2, 128, MR, 128, 1);
    gemm(B2, 128, g_m2t[i], 128, F(pb + 5), F3, nullptr, 0, MR, 128, 0);
    mean_l_kernel<<<dim3(MU), dim3(128), 0, stream>>>(F3, meanb);
    pack_concat_kernel<<<dim3(MR), dim3(128), 0, stream>>>(F2, meanb, B2);
    gemm(B2, 256, g_u1t[i], 256, F(pb + 7), nullptr, B1, 128, MR, 128, 1);
    gemm(B1, 128, g_u2t[i], 128, F(pb + 9), F1, nullptr, 0, MR, 128, 0);
    resid_ln_kernel<<<dim3(MR), dim3(128), 0, stream>>>(F1, F2, F(pb + 1), F(pb + 0),
                                                        F2, B1);
  }

  // --- attention aggregation over APs ---
  agg_score_kernel<<<dim3(MR / 256), dim3(256), 0, stream>>>(
      F2, F(P_AGG), F(P_AGG + 1), F(P_AGG + 2), F(P_AGG + 3), score, MR);
  agg_softmax_u_kernel<<<dim3(MU), dim3(128), 0, stream>>>(F2, score, u);

  // --- 2 pre-LN transformer layers on u (B,K,D) ---
  for (int t = 0; t < 2; ++t) {
    int pb = P_TF + 12 * t;
    resid_ln_kernel<<<dim3(MU), dim3(128), 0, stream>>>(u, nullptr, F(pb + 9), F(pb + 8),
                                                        nullptr, lnbf);
    gemm(lnbf, 128, t_qkvt[t], 128, F(pb + 3), qkv, nullptr, 0, MU, 384, 0);
    attention_kernel<<<dim3(256), dim3(32), 0, stream>>>(qkv, attnbf);
    gemm(attnbf, 128, t_wot[t], 128, F(pb + 2), tbuf, nullptr, 0, MU, 128, 0);
    add_kernel<<<dim3((MU * 128) / 256), dim3(256), 0, stream>>>(u, tbuf, MU * 128);
    resid_ln_kernel<<<dim3(MU), dim3(128), 0, stream>>>(u, nullptr, F(pb + 11), F(pb + 10),
                                                        nullptr, lnbf);
    gemm(lnbf, 128, t_f1t[t], 128, F(pb + 5), nullptr, ffhbf, 256, MU, 256, 1);
    gemm(ffhbf, 256, t_f2t[t], 256, F(pb + 7), tbuf, nullptr, 0, MU, 128, 0);
    add_kernel<<<dim3((MU * 128) / 256), dim3(256), 0, stream>>>(u, tbuf, MU * 128);
  }

  // --- head + combine -> (logits, s_global, s_combined) flat ---
  head_combine_kernel<<<dim3(8), dim3(256), 0, stream>>>(
      u, F(P_HEAD), F(P_HEAD + 1), F(P_HEAD + 2), F(P_HEAD + 3),
      F(P_ALPHA), s_glob, (float*)d_out);
}
